// TruthGPTSparseAttention_50964081934737
// MI455X (gfx1250) — matrix operand order repaired and checked by reading, not compiled, not run on hardware
//
#include <hip/hip_runtime.h>
#include <hip/hip_bf16.h>
#include <math.h>

// ---------------------------------------------------------------------------
// TruthGPT sparse attention, MI455X (gfx1250, wave32, WMMA).
//
// out = ( softmax( (QK^T/sqrt(128)) * strided_mask ) V ) Wo + bo
// mask zeroes scores (not -inf):
//   rows i%10!=0 : softmax uniform -> ctx = mean_j(V_j)
//   rows i%10==0 : ctx = (sum_win (e^s-1) V_j + sum_all V_j)
//                        / (sum_win (e^s-1) + S)
// Cost is dominated by 4x [4096x2048]x[2048x2048] GEMMs -> f16 WMMA, f32 acc.
// ---------------------------------------------------------------------------

typedef __attribute__((ext_vector_type(16))) _Float16 v16h;
typedef __attribute__((ext_vector_type(8)))  _Float16 v8h;
typedef __attribute__((ext_vector_type(4)))  _Float16 v4h;
typedef __attribute__((ext_vector_type(8)))  float    v8f;
typedef __attribute__((ext_vector_type(4)))  int      v4i;

#define S_LEN   2048
#define HID     2048
#define NHEADS  16
#define HDIM    128
#define BATCH   2
#define MROWS   (BATCH * S_LEN)   // 4096
#define STRIDE_ 10

// CDNA5 async global->LDS (ASYNCcnt-tracked) if the toolchain declares it.
#if __has_builtin(__builtin_amdgcn_global_load_async_to_lds_b128) && \
    __has_builtin(__builtin_amdgcn_s_wait_asynccnt)
#define USE_ASYNC_LDS 1
#else
#define USE_ASYNC_LDS 0
#endif

#if USE_ASYNC_LDS
// builtin signature (from compiler diagnostic): (v4i as1*, v4i as3*, imm, imm)
typedef __attribute__((address_space(1))) v4i GlobalV4i;
typedef __attribute__((address_space(3))) v4i LdsV4i;
#endif

// ---------------------------------------------------------------------------
// f32 -> f16 elementwise convert (4 elems / thread, coalesced b128 loads)
// ---------------------------------------------------------------------------
__global__ void cvt_f32_to_f16(const float* __restrict__ in,
                               _Float16* __restrict__ out, int n) {
  int i = (blockIdx.x * blockDim.x + threadIdx.x) * 4;
  if (i + 3 < n) {
    float4 x = *reinterpret_cast<const float4*>(in + i);
    v4h y;
    y.x = (_Float16)x.x; y.y = (_Float16)x.y;
    y.z = (_Float16)x.z; y.w = (_Float16)x.w;
    *reinterpret_cast<v4h*>(out + i) = y;
  }
}

// ---------------------------------------------------------------------------
// Transpose + convert: Wt[n][k] (f16) = W[k][n] (f32), both 2048x2048.
// ---------------------------------------------------------------------------
__global__ void transpose_cvt(const float* __restrict__ W,
                              _Float16* __restrict__ Wt) {
  __shared__ float tile[32][33];
  int tx = threadIdx.x, ty = threadIdx.y;
  int bx = blockIdx.x * 32, by = blockIdx.y * 32;
#pragma unroll
  for (int j = 0; j < 4; ++j) {
    int r = ty + j * 8;
    tile[r][tx] = W[(size_t)(by + r) * HID + (bx + tx)];
  }
  __syncthreads();
#pragma unroll
  for (int j = 0; j < 4; ++j) {
    int r = ty + j * 8;
    Wt[(size_t)(bx + r) * HID + (by + tx)] = (_Float16)tile[tx][r];
  }
}

// ---------------------------------------------------------------------------
// WMMA GEMM:  C[M,N] = A[M,K] * Bt[N,K]^T + bias[N]
// Block: 256 threads = 8 waves; block tile 256(M) x 128(N); BK=32.
// Wave (wm 0..3, wn 0..1) owns 64x64 = 4x4 WMMA subtiles (16 wmma / K-step).
// Double-buffered LDS, one barrier per K-step; async global->LDS when
// available, otherwise register-staged copy.
// ---------------------------------------------------------------------------
union F16Frag { v16h v; v8h h[2]; };

template <typename OutT>
__global__ __launch_bounds__(256)
void gemm_wmma_f16(const _Float16* __restrict__ A,
                   const _Float16* __restrict__ Bt,
                   const float* __restrict__ bias,
                   OutT* __restrict__ C,
                   int N, int K) {
  __shared__ __align__(16) _Float16 As[2][256][40];  // 40 KB
  __shared__ __align__(16) _Float16 Bs[2][128][40];  // 20 KB

  const int t    = threadIdx.x;
  const int lane = t & 31;
  const int wid  = t >> 5;
  const int wm   = wid & 3;   // 64-row strip
  const int wn   = wid >> 2;  // 64-col strip
  const int lr   = lane & 15;
  const int lh   = lane >> 4;

  const int bm = blockIdx.y * 256;
  const int bn = blockIdx.x * 128;

  // tile-copy assignment: b128 chunks (8 halves); A: 1024 chunks, B: 512
  const int grow = t >> 2;        // 0..63
  const int gcol = (t & 3) * 8;   // 0,8,16,24

  const int nsteps = K / 32;

  v8f acc[4][4];
  const v8f vzero = {0.f, 0.f, 0.f, 0.f, 0.f, 0.f, 0.f, 0.f};
#pragma unroll
  for (int i = 0; i < 4; ++i)
#pragma unroll
    for (int j = 0; j < 4; ++j) acc[i][j] = vzero;

  auto compute = [&](int cur) {
    F16Frag bfrag[4];
#pragma unroll
    for (int ns = 0; ns < 4; ++ns) {
      const _Float16* p = &Bs[cur][wn * 64 + ns * 16 + lr][lh * 16];
      bfrag[ns].h[0] = *reinterpret_cast<const v8h*>(p);
      bfrag[ns].h[1] = *reinterpret_cast<const v8h*>(p + 8);
    }
#pragma unroll
    for (int ms = 0; ms < 4; ++ms) {
      F16Frag a;
      const _Float16* p = &As[cur][wm * 64 + ms * 16 + lr][lh * 16];
      a.h[0] = *reinterpret_cast<const v8h*>(p);
      a.h[1] = *reinterpret_cast<const v8h*>(p + 8);
#pragma unroll
      for (int ns = 0; ns < 4; ++ns)
        acc[ms][ns] = __builtin_amdgcn_wmma_f32_16x16x32_f16(
            false, a.v, false, bfrag[ns].v, (short)0, acc[ms][ns],
            false, false);
    }
  };

#if USE_ASYNC_LDS
  auto copy_tile = [&](int buf, int k0) {
#pragma unroll
    for (int j = 0; j < 4; ++j) {  // A: 256 rows
      int row = grow + 64 * j;
      __builtin_amdgcn_global_load_async_to_lds_b128(
          (GlobalV4i*)(A + (size_t)(bm + row) * K + k0 + gcol),
          (LdsV4i*)&As[buf][row][gcol], 0, 0);
    }
#pragma unroll
    for (int j = 0; j < 2; ++j) {  // B: 128 rows
      int row = grow + 64 * j;
      __builtin_amdgcn_global_load_async_to_lds_b128(
          (GlobalV4i*)(Bt + (size_t)(bn + row) * K + k0 + gcol),
          (LdsV4i*)&Bs[buf][row][gcol], 0, 0);
    }
  };

  copy_tile(0, 0);
  __builtin_amdgcn_s_wait_asynccnt(0);
  __syncthreads();
  for (int i = 0; i < nsteps; ++i) {
    const int cur = i & 1;
    if (i + 1 < nsteps) copy_tile(1 - cur, (i + 1) * 32);  // overlaps compute
    compute(cur);
    if (i + 1 < nsteps) {
      __builtin_amdgcn_s_wait_asynccnt(0);
      __syncthreads();
    }
  }
#else
  v8h ra[4], rb[2];
  auto gload = [&](int k0) {
#pragma unroll
    for (int j = 0; j < 4; ++j)
      ra[j] = *reinterpret_cast<const v8h*>(
          A + (size_t)(bm + grow + 64 * j) * K + k0 + gcol);
#pragma unroll
    for (int j = 0; j < 2; ++j)
      rb[j] = *reinterpret_cast<const v8h*>(
          Bt + (size_t)(bn + grow + 64 * j) * K + k0 + gcol);
  };

  gload(0);
  for (int i = 0; i < nsteps; ++i) {
    const int cur = i & 1;
#pragma unroll
    for (int j = 0; j < 4; ++j)
      *reinterpret_cast<v8h*>(&As[cur][grow + 64 * j][gcol]) = ra[j];
#pragma unroll
    for (int j = 0; j < 2; ++j)
      *reinterpret_cast<v8h*>(&Bs[cur][grow + 64 * j][gcol]) = rb[j];
    if (i + 1 < nsteps) gload((i + 1) * 32);  // in flight during compute
    __syncthreads();
    compute(cur);
  }
#endif

  // ---- epilogue: D layout -> (row = r + 8*lane_hi, col = lane_lo) ----
#pragma unroll
  for (int ms = 0; ms < 4; ++ms)
#pragma unroll
    for (int ns = 0; ns < 4; ++ns) {
      int gn = bn + wn * 64 + ns * 16 + lr;
      float bias_n = bias[gn];
      int rowbase = bm + wm * 64 + ms * 16 + lh * 8;
      v8f c = acc[ms][ns];
#pragma unroll
      for (int r = 0; r < 8; ++r)
        C[(size_t)(rowbase + r) * N + gn] = (OutT)(c[r] + bias_n);
    }
}

// ---------------------------------------------------------------------------
// Vsum[b,h,d] = sum_j V[b,j,h*128+d] -- two-stage deterministic reduction
// ---------------------------------------------------------------------------
__global__ void vsum_stage1(const _Float16* __restrict__ Vh,
                            float* __restrict__ partial) {
  int bh = blockIdx.x;          // 0..31  (b*16+h)
  int jc = blockIdx.y;          // 0..15  (128-row chunk)
  int b = bh >> 4, h = bh & 15;
  int d = threadIdx.x;          // 0..127
  size_t base = (size_t)b * S_LEN * HID + (size_t)h * HDIM + d +
                (size_t)(jc * 128) * HID;
  float acc = 0.f;
#pragma unroll 8
  for (int j = 0; j < 128; ++j)
    acc += (float)Vh[base + (size_t)j * HID];
  partial[((size_t)jc * 32 + bh) * HDIM + d] = acc;
}

__global__ void vsum_stage2(const float* __restrict__ partial,
                            float* __restrict__ Vsum) {
  int bh = blockIdx.x;
  int d = threadIdx.x;
  float acc = 0.f;
#pragma unroll
  for (int jc = 0; jc < 16; ++jc)
    acc += partial[((size_t)jc * 32 + bh) * HDIM + d];
  Vsum[bh * HDIM + d] = acc;
}

// ---------------------------------------------------------------------------
// Sparse strided attention (closed form). One block (128 thr) per
// (window group g, head h, batch b); covers rows [10g, 10g+10).
// ---------------------------------------------------------------------------
__global__ void sparse_attn_kernel(const _Float16* __restrict__ Qh,
                                   const _Float16* __restrict__ Kh,
                                   const _Float16* __restrict__ Vh,
                                   const float* __restrict__ Vsum,
                                   _Float16* __restrict__ Ctx) {
  const int g = blockIdx.x, h = blockIdx.y, b = blockIdx.z;
  const int i0 = g * STRIDE_;
  const int d = threadIdx.x;
  __shared__ float red[128];
  __shared__ float wsh[STRIDE_];

  const size_t rowbase = (size_t)b * S_LEN * HID + (size_t)h * HDIM + d;
  const int nwin = min(STRIDE_, S_LEN - i0);
  const float scale = 0.08838834764831845f;  // 1/sqrt(128)

  const float q = (float)Qh[rowbase + (size_t)i0 * HID];
  for (int jj = 0; jj < nwin; ++jj) {
    red[d] = q * (float)Kh[rowbase + (size_t)(i0 + jj) * HID];
    __syncthreads();
#pragma unroll
    for (int s = 64; s > 0; s >>= 1) {
      if (d < s) red[d] += red[d + s];
      __syncthreads();
    }
    if (d == 0) wsh[jj] = expf(red[0] * scale) - 1.0f;
    __syncthreads();
  }

  const float vs = Vsum[(b * NHEADS + h) * HDIM + d];
  float wsum = 0.f, vacc = vs;
  for (int jj = 0; jj < nwin; ++jj) {
    float w = wsh[jj];
    wsum += w;
    vacc += w * (float)Vh[rowbase + (size_t)(i0 + jj) * HID];
  }
  Ctx[rowbase + (size_t)i0 * HID] = (_Float16)(vacc / (wsum + (float)S_LEN));

  const float mean = vs * (1.0f / (float)S_LEN);
  for (int r = 1; r < STRIDE_; ++r) {
    int row = i0 + r;
    if (row < S_LEN)
      Ctx[rowbase + (size_t)row * HID] = (_Float16)mean;
  }
}

// ---------------------------------------------------------------------------
// Host-side orchestration
// ---------------------------------------------------------------------------
extern "C" void kernel_launch(void* const* d_in, const int* in_sizes, int n_in,
                              void* d_out, int out_size, void* d_ws, size_t ws_size,
                              hipStream_t stream) {
  const float* X  = (const float*)d_in[0];
  const float* wq = (const float*)d_in[1];
  const float* bq = (const float*)d_in[2];
  const float* wk = (const float*)d_in[3];
  const float* bk = (const float*)d_in[4];
  const float* wv = (const float*)d_in[5];
  const float* bv = (const float*)d_in[6];
  const float* wo = (const float*)d_in[7];
  const float* bo = (const float*)d_in[8];
  float* out = (float*)d_out;

  char* ws = (char*)d_ws;
  const size_t SZ_ACT = (size_t)MROWS * HID * sizeof(_Float16);  // 16 MB
  const size_t SZ_W   = (size_t)HID * HID * sizeof(_Float16);    //  8 MB
  size_t off = 0;
  _Float16* Xh   = (_Float16*)(ws + off); off += SZ_ACT;
  _Float16* Wqt  = (_Float16*)(ws + off); off += SZ_W;
  _Float16* Wkt  = (_Float16*)(ws + off); off += SZ_W;
  _Float16* Wvt  = (_Float16*)(ws + off); off += SZ_W;
  _Float16* Wot  = (_Float16*)(ws + off); off += SZ_W;
  _Float16* Qh   = (_Float16*)(ws + off); off += SZ_ACT;
  _Float16* Kh   = (_Float16*)(ws + off); off += SZ_ACT;
  _Float16* Vh   = (_Float16*)(ws + off); off += SZ_ACT;
  _Float16* Ctxh = (_Float16*)(ws + off); off += SZ_ACT;
  float*    Vsum = (float*)(ws + off);
  off += (size_t)BATCH * NHEADS * HDIM * sizeof(float);
  float*    Vpart = (float*)(ws + off);
  off += (size_t)16 * 32 * HDIM * sizeof(float);

  // 1) convert inputs
  {
    int n = MROWS * HID;
    cvt_f32_to_f16<<<n / (256 * 4), 256, 0, stream>>>(X, Xh, n);
  }
  {
    dim3 blk(32, 8), grd(HID / 32, HID / 32);
    transpose_cvt<<<grd, blk, 0, stream>>>(wq, Wqt);
    transpose_cvt<<<grd, blk, 0, stream>>>(wk, Wkt);
    transpose_cvt<<<grd, blk, 0, stream>>>(wv, Wvt);
    transpose_cvt<<<grd, blk, 0, stream>>>(wo, Wot);
  }

  // 2) Q/K/V projections (f16 out)
  dim3 ggrid(HID / 128, MROWS / 256);  // (16, 16)
  gemm_wmma_f16<_Float16><<<ggrid, 256, 0, stream>>>(Xh, Wqt, bq, Qh, HID, HID);
  gemm_wmma_f16<_Float16><<<ggrid, 256, 0, stream>>>(Xh, Wkt, bk, Kh, HID, HID);
  gemm_wmma_f16<_Float16><<<ggrid, 256, 0, stream>>>(Xh, Wvt, bv, Vh, HID, HID);

  // 3) attention (closed form for zero-masked strided softmax)
  {
    dim3 vgrid(BATCH * NHEADS, 16);
    vsum_stage1<<<vgrid, HDIM, 0, stream>>>(Vh, Vpart);
    vsum_stage2<<<BATCH * NHEADS, HDIM, 0, stream>>>(Vpart, Vsum);
    dim3 agrid((S_LEN + STRIDE_ - 1) / STRIDE_, NHEADS, BATCH);  // (205,16,2)
    sparse_attn_kernel<<<agrid, HDIM, 0, stream>>>(Qh, Kh, Vh, Vsum, Ctxh);
  }

  // 4) output projection (f32 out -> d_out)
  gemm_wmma_f16<float><<<ggrid, 256, 0, stream>>>(Ctxh, Wot, bo, out, HID, HID);
}
